// OptimizedGAT_51196010169100
// MI455X (gfx1250) — compile-verified, matrix operand order checked by reading, not run on hardware
//
#include <hip/hip_runtime.h>
#include <hip/hip_bf16.h>

typedef __attribute__((ext_vector_type(16))) __bf16 v16bf;
typedef __attribute__((ext_vector_type(8)))  __bf16 v8bf;
typedef __attribute__((ext_vector_type(8)))  float  v8f;

#define N_NODES  100000
#define N_EDGES  1600000
#define EL_EDGES (N_EDGES + N_NODES)   // with self-loops
#define NH       4                     // heads
#define NC       32                    // per-head dim
#define HC       128                   // NH*NC
#define NG       64                    // graphs
#define NDOUT    16
#define NEG_SLOPE 0.2f
#define BN_EPS   1e-5f
#define SM_EPS   1e-16f

// ---------------------------------------------------------------------------
// GEMM: Y[rows x 128] = X[rows x 128] @ W[128 x 128]  (bf16 WMMA, f32 accum)
// Block: 256 thr = 8 waves. Block owns 16 rows; wave w owns cols [16w,16w+16).
// A tile and transposed W are staged in LDS as bf16 so every fragment load is
// a 16-byte ds_load_b128 (two contiguous 8-element K runs per lane).
// ---------------------------------------------------------------------------
__global__ void gat_gemm_wmma(const float* __restrict__ X,
                              const float* __restrict__ W,
                              float* __restrict__ Y) {
  __shared__ __bf16 Wt[HC * HC];      // transposed: Wt[n*128 + k]  (32 KB)
  __shared__ __bf16 At[16 * HC];      // A tile bf16: At[m*128 + k] ( 4 KB)

  const int t    = threadIdx.x;
  const int row0 = blockIdx.x * 16;

  // ---- stage W transposed (coalesced global float4 reads) ----
  for (int i = t; i < HC * HC / 4; i += 256) {
    const int k = i >> 5;             // W row
    const int n = (i & 31) * 4;       // W col (4 consecutive)
    const float4 f = *(const float4*)(W + (size_t)k * HC + n);
    Wt[(n + 0) * HC + k] = (__bf16)f.x;
    Wt[(n + 1) * HC + k] = (__bf16)f.y;
    Wt[(n + 2) * HC + k] = (__bf16)f.z;
    Wt[(n + 3) * HC + k] = (__bf16)f.w;
  }

  // ---- stage A tile: 16 rows x 128 cols, 8 floats per thread ----
  {
    const int m = t >> 4;
    const int k = (t & 15) * 8;
    const float* xp = X + (size_t)(row0 + m) * HC + k;
    const float4 f0 = *(const float4*)(xp);
    const float4 f1 = *(const float4*)(xp + 4);
    v8bf av;
    av[0] = (__bf16)f0.x; av[1] = (__bf16)f0.y;
    av[2] = (__bf16)f0.z; av[3] = (__bf16)f0.w;
    av[4] = (__bf16)f1.x; av[5] = (__bf16)f1.y;
    av[6] = (__bf16)f1.z; av[7] = (__bf16)f1.w;
    *(v8bf*)&At[m * HC + k] = av;     // ds_store_b128, conflict-free
  }
  __syncthreads();

  const int wave = t >> 5;
  const int lane = t & 31;
  const int half = lane >> 4;         // 0 or 1
  const int m16  = lane & 15;
  const int ncol = wave * 16 + m16;

  v8f acc = {};
  #pragma unroll
  for (int kc = 0; kc < 4; ++kc) {    // K = 4 * 32
    // A fragment: lane = row m16; K runs [kc*32+half*8, +8) and +16
    const v8bf* ap = (const v8bf*)&At[m16 * HC + kc * 32 + half * 8];
    const v8bf alo = ap[0];
    const v8bf ahi = ap[2];           // +16 bf16 elements
    const v16bf a = __builtin_shufflevector(alo, ahi,
        0, 1, 2, 3, 4, 5, 6, 7, 8, 9, 10, 11, 12, 13, 14, 15);

    // B fragment: lane = col ncol; identical K-run pattern (transposed W)
    const v8bf* bp = (const v8bf*)&Wt[ncol * HC + kc * 32 + half * 8];
    const v8bf blo = bp[0];
    const v8bf bhi = bp[2];
    const v16bf b = __builtin_shufflevector(blo, bhi,
        0, 1, 2, 3, 4, 5, 6, 7, 8, 9, 10, 11, 12, 13, 14, 15);

    acc = __builtin_amdgcn_wmma_f32_16x16x32_bf16(
        /*neg_a=*/false, a, /*neg_b=*/false, b,
        /*c_mod=*/(short)0, acc, /*reuse_a=*/false, /*reuse_b=*/false);
  }

  // C/D layout: VGPR v, lanes 0-15 -> M=v, lanes 16-31 -> M=v+8; N = lane&15
  #pragma unroll
  for (int v = 0; v < 8; ++v)
    Y[(size_t)(row0 + v + 8 * half) * HC + ncol] = acc[v];
}

// ---------------------------------------------------------------------------
// Attention scores: a_src/a_dst[n,h] = sum_c h[n,h,c] * att_{src,dst}[h,c]
// ---------------------------------------------------------------------------
__global__ void gat_scores(const float* __restrict__ hlin,
                           const float* __restrict__ att_src,
                           const float* __restrict__ att_dst,
                           float* __restrict__ aS, float* __restrict__ aD) {
  const int gid = blockIdx.x * blockDim.x + threadIdx.x;
  if (gid >= N_NODES * NH) return;
  const int n = gid >> 2, hh = gid & 3;
  const float* hp = hlin + (size_t)n * HC + hh * NC;
  const float* as = att_src + hh * NC;
  const float* ad = att_dst + hh * NC;
  float ssum = 0.f, dsum = 0.f;
  #pragma unroll 8
  for (int c = 0; c < NC; ++c) {
    const float hv = hp[c];
    ssum += hv * as[c];
    dsum += hv * ad[c];
  }
  aS[gid] = ssum;
  aD[gid] = dsum;
}

__device__ __forceinline__ unsigned enc_f32(float f) {
  unsigned u = __float_as_uint(f);
  return (u & 0x80000000u) ? ~u : (u | 0x80000000u);
}
__device__ __forceinline__ float dec_f32(unsigned u) {
  if (u == 0u) return 0.0f;                       // no edges -> ref uses 0
  return (u & 0x80000000u) ? __uint_as_float(u & 0x7FFFFFFFu)
                           : __uint_as_float(~u);
}

// pass 1: e = leaky_relu(aS[src]+aD[dst]); store e; segment-max into mEnc[dst]
__global__ void gat_edge_max(const float* __restrict__ aS,
                             const float* __restrict__ aD,
                             const int* __restrict__ ei,
                             float* __restrict__ p,
                             unsigned* __restrict__ mEnc) {
  const int gid = blockIdx.x * blockDim.x + threadIdx.x;
  if (gid >= EL_EDGES * NH) return;
  const int edge = gid >> 2, hh = gid & 3;
  int src, dst;
  if (edge < N_EDGES) { src = ei[edge]; dst = ei[N_EDGES + edge]; }
  else                { src = dst = edge - N_EDGES; }
  float e = aS[src * NH + hh] + aD[dst * NH + hh];
  e = (e > 0.f) ? e : NEG_SLOPE * e;
  p[gid] = e;
  atomicMax(&mEnc[dst * NH + hh], enc_f32(e));
}

// pass 2: p = exp(e - m[dst]); segment-sum into s[dst]
__global__ void gat_edge_exp(const int* __restrict__ ei,
                             const unsigned* __restrict__ mEnc,
                             float* __restrict__ p,
                             float* __restrict__ s) {
  const int gid = blockIdx.x * blockDim.x + threadIdx.x;
  if (gid >= EL_EDGES * NH) return;
  const int edge = gid >> 2, hh = gid & 3;
  int dst;
  if (edge < N_EDGES) dst = ei[N_EDGES + edge];
  else                dst = edge - N_EDGES;
  const float m  = dec_f32(mEnc[dst * NH + hh]);
  const float pe = __expf(p[gid] - m);
  p[gid] = pe;
  atomicAdd(&s[dst * NH + hh], pe);
}

// aggregation: out[dst] += h[src] * alpha   (one wave per edge, 4 ch / lane)
__global__ void gat_aggregate(const float* __restrict__ hlin,
                              const float* __restrict__ p,
                              const float* __restrict__ s,
                              const int* __restrict__ ei,
                              float* __restrict__ out) {
  const long long gid = (long long)blockIdx.x * blockDim.x + threadIdx.x;
  const int edge = (int)(gid >> 5);
  const int lane = (int)(gid & 31);
  if (edge >= EL_EDGES) return;
  int src, dst;
  if (edge < N_EDGES) { src = ei[edge]; dst = ei[N_EDGES + edge]; }
  else                { src = dst = edge - N_EDGES; }
  const int c0 = lane * 4;
  const int hh = c0 >> 5;
  const float alpha = p[(size_t)edge * NH + hh] / (s[dst * NH + hh] + SM_EPS);
  const float4 hv = *(const float4*)(hlin + (size_t)src * HC + c0);
  float* o = out + (size_t)dst * HC + c0;
  atomicAdd(o + 0, hv.x * alpha);
  atomicAdd(o + 1, hv.y * alpha);
  atomicAdd(o + 2, hv.z * alpha);
  atomicAdd(o + 3, hv.w * alpha);
}

// bias + batchnorm(eval) + ELU, in place
__global__ void gat_bn_elu(float* __restrict__ h,
                           const float* __restrict__ bias,
                           const float* __restrict__ g,
                           const float* __restrict__ b,
                           const float* __restrict__ m,
                           const float* __restrict__ v) {
  const int gid = blockIdx.x * blockDim.x + threadIdx.x;
  if (gid >= N_NODES * HC) return;
  const int c = gid & (HC - 1);
  float x = h[gid] + bias[c];
  x = (x - m[c]) * rsqrtf(v[c] + BN_EPS) * g[c] + b[c];
  h[gid] = (x > 0.f) ? x : (__expf(x) - 1.0f);
}

// mean-pool accumulation
__global__ void gat_pool(const float* __restrict__ h,
                         const int* __restrict__ batch,
                         float* __restrict__ sums,
                         float* __restrict__ cnt) {
  const int gid = blockIdx.x * blockDim.x + threadIdx.x;
  if (gid >= N_NODES * HC) return;
  const int n = gid >> 7, c = gid & (HC - 1);
  const int gr = batch[n];
  atomicAdd(&sums[gr * HC + c], h[gid]);
  if (c == 0) atomicAdd(&cnt[gr], 1.0f);
}

// final linear: out[g,o] = sum_c (sums[g,c]/max(cnt,1)) * Wf[c,o] + bf[o]
__global__ void gat_final(const float* __restrict__ sums,
                          const float* __restrict__ cnt,
                          const float* __restrict__ Wf,
                          const float* __restrict__ bf,
                          float* __restrict__ out) {
  const int tid = blockIdx.x * blockDim.x + threadIdx.x;
  if (tid >= NG * NDOUT) return;
  const int g = tid >> 4, o = tid & 15;
  const float inv = 1.0f / fmaxf(cnt[g], 1.0f);
  float acc = bf[o];
  #pragma unroll 8
  for (int c = 0; c < HC; ++c)
    acc += sums[g * HC + c] * inv * Wf[c * NDOUT + o];
  out[tid] = acc;
}

// ---------------------------------------------------------------------------
extern "C" void kernel_launch(void* const* d_in, const int* in_sizes, int n_in,
                              void* d_out, int out_size, void* d_ws, size_t ws_size,
                              hipStream_t stream) {
  const float* x        = (const float*)d_in[0];
  const int*   ei       = (const int*)  d_in[1];
  const int*   batch    = (const int*)  d_in[2];
  const float* W0       = (const float*)d_in[3];
  const float* att_src0 = (const float*)d_in[4];
  const float* att_dst0 = (const float*)d_in[5];
  const float* b0       = (const float*)d_in[6];
  const float* W1       = (const float*)d_in[7];
  const float* att_src1 = (const float*)d_in[8];
  const float* att_dst1 = (const float*)d_in[9];
  const float* b1       = (const float*)d_in[10];
  const float* bn_g0 = (const float*)d_in[11];
  const float* bn_b0 = (const float*)d_in[12];
  const float* bn_m0 = (const float*)d_in[13];
  const float* bn_v0 = (const float*)d_in[14];
  const float* bn_g1 = (const float*)d_in[15];
  const float* bn_b1 = (const float*)d_in[16];
  const float* bn_m1 = (const float*)d_in[17];
  const float* bn_v1 = (const float*)d_in[18];
  const float* Wf = (const float*)d_in[19];
  const float* bf = (const float*)d_in[20];
  float* out = (float*)d_out;

  // workspace carve-up (all sizes 256B aligned)
  char* w = (char*)d_ws;
  auto take = [&](size_t bytes) {
    void* ptr = (void*)w;
    w += (bytes + 255) & ~(size_t)255;
    return ptr;
  };
  float*    bufA = (float*)   take((size_t)N_NODES * HC * 4);
  float*    bufB = (float*)   take((size_t)N_NODES * HC * 4);
  float*    aS   = (float*)   take((size_t)N_NODES * NH * 4);
  float*    aD   = (float*)   take((size_t)N_NODES * NH * 4);
  unsigned* mEnc = (unsigned*)take((size_t)N_NODES * NH * 4);
  float*    sSum = (float*)   take((size_t)N_NODES * NH * 4);
  float*    pBuf = (float*)   take((size_t)EL_EDGES * NH * 4);
  float*    sums = (float*)   take((size_t)NG * HC * 4);
  float*    cnt  = (float*)   take((size_t)NG * 4);

  const int gGemm  = N_NODES / 16;                         // 6250
  const int gScore = (N_NODES * NH + 255) / 256;
  const int gEdge  = (EL_EDGES * NH + 255) / 256;
  const int gAgg   = (int)(((long long)EL_EDGES * 32 + 255) / 256);
  const int gElem  = (N_NODES * HC + 255) / 256;

  const float* layerW[2]   = {W0, W1};
  const float* layerAS[2]  = {att_src0, att_src1};
  const float* layerAD[2]  = {att_dst0, att_dst1};
  const float* layerB[2]   = {b0, b1};
  const float* layerBNg[2] = {bn_g0, bn_g1};
  const float* layerBNb[2] = {bn_b0, bn_b1};
  const float* layerBNm[2] = {bn_m0, bn_m1};
  const float* layerBNv[2] = {bn_v0, bn_v1};

  for (int layer = 0; layer < 2; ++layer) {
    const float* xin = (layer == 0) ? x : bufA;

    gat_gemm_wmma<<<gGemm, 256, 0, stream>>>(xin, layerW[layer], bufB);
    gat_scores<<<gScore, 256, 0, stream>>>(bufB, layerAS[layer], layerAD[layer], aS, aD);

    hipMemsetAsync(mEnc, 0, (size_t)N_NODES * NH * 4, stream);
    hipMemsetAsync(sSum, 0, (size_t)N_NODES * NH * 4, stream);
    gat_edge_max<<<gEdge, 256, 0, stream>>>(aS, aD, ei, pBuf, mEnc);
    gat_edge_exp<<<gEdge, 256, 0, stream>>>(ei, mEnc, pBuf, sSum);

    hipMemsetAsync(bufA, 0, (size_t)N_NODES * HC * 4, stream);
    gat_aggregate<<<gAgg, 256, 0, stream>>>(bufB, pBuf, sSum, ei, bufA);

    gat_bn_elu<<<gElem, 256, 0, stream>>>(bufA, layerB[layer],
                                          layerBNg[layer], layerBNb[layer],
                                          layerBNm[layer], layerBNv[layer]);
  }

  hipMemsetAsync(sums, 0, (size_t)NG * HC * 4, stream);
  hipMemsetAsync(cnt, 0, (size_t)NG * 4, stream);
  gat_pool<<<gElem, 256, 0, stream>>>(bufA, batch, sums, cnt);
  gat_final<<<(NG * NDOUT + 255) / 256, 256, 0, stream>>>(sums, cnt, Wf, bf, out);
}
